// Decoder_64424509440568
// MI455X (gfx1250) — matrix operand order repaired and checked by reading, not compiled, not run on hardware
//
#include <hip/hip_runtime.h>

// ---------------------------------------------------------------------------
// Problem constants (from the reference)
// ---------------------------------------------------------------------------
constexpr int kV = 32000;   // vocab
constexpr int kD = 1024;    // embedding dim (== H)
constexpr int kH = 1024;    // hidden
constexpr int kL = 4;       // layers
constexpr int kB = 32;      // batch
constexpr int kT = 128;     // time steps
constexpr int kG = 4 * kH;  // gate width (4096)
constexpr int kM = kT * kB; // flattened (t,b) rows = 4096

// ---------------------------------------------------------------------------
// CDNA5 WMMA types / helpers (wave32, V_WMMA_F32_16X16X32_BF16)
// ---------------------------------------------------------------------------
typedef __attribute__((ext_vector_type(16))) __bf16 v16bf;
typedef __attribute__((ext_vector_type(8)))  __bf16 v8bf;
typedef __attribute__((ext_vector_type(8)))  float  v8f;
typedef __attribute__((ext_vector_type(4)))  float  v4f;

__device__ __forceinline__ v8f wmma_bf16(v16bf a, v16bf b, v8f c) {
  // 8 args: (neg_a, A, neg_b, B, c_mod, C, reuse_a, reuse_b)
  return __builtin_amdgcn_wmma_f32_16x16x32_bf16(false, a, false, b,
                                                 (short)0, c, false, false);
}

// A fragment (16x32, bf16 source, row-major, stride lda).
// Lane l (l<16): row mBase+l, K = k0..k0+15 contiguous
// Lane l+16   : row mBase+l, K = k0+16..k0+31
__device__ __forceinline__ v16bf load_a_bf16(const __bf16* A, int lda,
                                             int mBase, int k0, int lane) {
  int r = lane & 15, h = (lane >> 4) & 1;
  const __bf16* p = A + (size_t)(mBase + r) * lda + k0 + h * 16;
  v8bf lo = *(const v8bf*)(p);
  v8bf hi = *(const v8bf*)(p + 8);
  v16bf o;
#pragma unroll
  for (int i = 0; i < 8; ++i) { o[i] = lo[i]; o[8 + i] = hi[i]; }
  return o;
}

// B fragment (32x16) built from a ROW-MAJOR weight matrix W[N][K]:
// B[k][n] = W[n][k].  Lane n (n<16): col nBase+n, K = k0..k0+15 contiguous.
// Lane n+16: col nBase+n, K = k0+16..k0+31.
__device__ __forceinline__ v16bf load_b_wrow(const __bf16* W, int ldw,
                                             int nBase, int k0, int lane) {
  int c = lane & 15, h = (lane >> 4) & 1;
  const __bf16* p = W + (size_t)(nBase + c) * ldw + k0 + h * 16;
  v8bf b0 = *(const v8bf*)(p);
  v8bf b1 = *(const v8bf*)(p + 8);
  v16bf o;
#pragma unroll
  for (int i = 0; i < 8; ++i) { o[i] = b0[i]; o[8 + i] = b1[i]; }
  return o;
}

__device__ __forceinline__ float sigmoidf(float x) {
  return 1.0f / (1.0f + __expf(-x));
}

// ---------------------------------------------------------------------------
// Grid-wide barrier (generation counter). d_ws-backed, memset to 0 before each
// persistent kernel launch.
// ---------------------------------------------------------------------------
__device__ __forceinline__ unsigned grid_barrier(unsigned* count, unsigned* gen,
                                                 unsigned nblocks, unsigned myGen) {
  __syncthreads();
  if (threadIdx.x == 0) {
    __threadfence();
    unsigned arrived = __hip_atomic_fetch_add(count, 1u, __ATOMIC_ACQ_REL,
                                              __HIP_MEMORY_SCOPE_AGENT);
    if (arrived == nblocks - 1) {
      __hip_atomic_store(count, 0u, __ATOMIC_RELAXED, __HIP_MEMORY_SCOPE_AGENT);
      __hip_atomic_fetch_add(gen, 1u, __ATOMIC_RELEASE, __HIP_MEMORY_SCOPE_AGENT);
    } else {
      while (__hip_atomic_load(gen, __ATOMIC_ACQUIRE,
                               __HIP_MEMORY_SCOPE_AGENT) <= myGen) {
        __builtin_amdgcn_s_sleep(2);
      }
    }
  }
  __syncthreads();
  return myGen + 1;
}

// ---------------------------------------------------------------------------
// Kernel 1: f32 -> bf16 weight conversion (one-time; weights then L2-resident)
// ---------------------------------------------------------------------------
__global__ void cvt_bf16_kernel(const float* __restrict__ in,
                                __bf16* __restrict__ out, size_t n) {
  size_t i = (size_t)blockIdx.x * blockDim.x + threadIdx.x;
  if (i < n) out[i] = (__bf16)in[i];
}

// ---------------------------------------------------------------------------
// Kernel 2: embedding gather -> x0 [T,B,D] f32 + bf16 shadow for GEMM A-side
// ---------------------------------------------------------------------------
__global__ void embed_kernel(const int* __restrict__ dec,
                             const float* __restrict__ emb,
                             float* __restrict__ x0,
                             __bf16* __restrict__ x0bf) {
  int blk = blockIdx.x;              // blk = t*B + b
  int t = blk >> 5, b = blk & 31;    // B == 32
  int tok = dec[b * kT + t];
  const float* src = emb + (size_t)tok * kD;
  float* dst = x0 + (size_t)blk * kD;
  __bf16* dstb = x0bf + (size_t)blk * kD;
  int d = threadIdx.x * 4;           // 256 threads * 4 floats = 1024
  v4f v = *(const v4f*)(src + d);
  *(v4f*)(dst + d) = v;
#pragma unroll
  for (int i = 0; i < 4; ++i) dstb[d + i] = (__bf16)v[i];
}

// ---------------------------------------------------------------------------
// Kernel 3: per-layer input-gate GEMM (parallel over all T):
//   Gin[m, n] = Xbf[m,:] @ W_ih[l][n,:] + b_ih[n] + b_hh[n],  m = t*B+b
// M=4096, N=4096, K=1024. Block = 4 waves; block tile 16(M) x 256(N);
// wave tile 16 x 64 (4 WMMA accumulators, A fragment reused 4x, pure bf16).
// ---------------------------------------------------------------------------
__global__ void __launch_bounds__(128)
gates_gemm_kernel(const __bf16* __restrict__ Xbf, const __bf16* __restrict__ W,
                  const float* __restrict__ bih, const float* __restrict__ bhh,
                  float* __restrict__ Gout) {
  int lane = threadIdx.x & 31;
  int wave = threadIdx.x >> 5;                 // 0..3
  int mBase = blockIdx.x * 16;
  int nBase = blockIdx.y * 256 + wave * 64;

  v8f acc[4] = {v8f{}, v8f{}, v8f{}, v8f{}};
  for (int k0 = 0; k0 < kH; k0 += 32) {
    v16bf a = load_a_bf16(Xbf, kH, mBase, k0, lane);
    __builtin_prefetch(W + (size_t)(nBase + (lane & 15)) * kH + k0 + 256, 0, 1);
#pragma unroll
    for (int j = 0; j < 4; ++j) {
      v16bf b = load_b_wrow(W, kH, nBase + j * 16, k0, lane);
      acc[j] = wmma_bf16(a, b, acc[j]);
    }
  }
  int col0 = lane & 15;
  int rbase = mBase + ((lane >> 4) & 1) * 8;
#pragma unroll
  for (int j = 0; j < 4; ++j) {
    int n = nBase + j * 16 + col0;
    float bias = bih[n] + bhh[n];
#pragma unroll
    for (int vi = 0; vi < 8; ++vi)
      Gout[(size_t)(rbase + vi) * kG + n] = acc[j][vi] + bias;
  }
}

// ---------------------------------------------------------------------------
// Kernel 4: persistent per-layer LSTM recurrence.
// Grid = 32 workgroups; workgroup wg owns hidden units [wg*32, wg*32+32) so
// all four gate columns (i,f,g,o) of a unit land in-workgroup. Per step:
// Z = h_{t-1} @ W_hh^T for the 8 owned WMMA col-tiles (8 waves x 2 row-tiles),
// stage Z through LDS, fused LSTM cell epilogue, residual write (f32 + bf16
// shadow), publish bf16 h into a double buffer, grid barrier.
// Cell state c lives in registers for all 128 steps.
// ---------------------------------------------------------------------------
#define NBLK 32
__global__ void __launch_bounds__(256)
lstm_layer_kernel(const float* __restrict__ Gin,   // [T, B, 4H]
                  const __bf16* __restrict__ Whh,  // [4H, H] bf16 row-major
                  __bf16* __restrict__ hbuf,       // 2 x [B, H] bf16 ping-pong
                  const float* __restrict__ h0,    // [B, H]
                  const float* __restrict__ c0,    // [B, H]
                  const float* __restrict__ x_in,  // [T, B, H] f32
                  float* __restrict__ x_out,       // [T, B, H] f32
                  __bf16* __restrict__ x_out_bf,   // [T, B, H] bf16 shadow
                  float* __restrict__ hT,          // [B, H]
                  float* __restrict__ cT,          // [B, H]
                  unsigned* bar_count, unsigned* bar_gen) {
  __shared__ float zsm[kB][128];                   // 16 KB: Z slice [batch][4*32]
  const int lane = threadIdx.x & 31;
  const int wave = threadIdx.x >> 5;               // 0..7
  const int wg = blockIdx.x;                       // 0..31
  const int jbase = wg * 32;                       // first owned hidden unit

  // Publish initial h (buffer 0) for the owned slice; keep c in registers.
  for (int i = threadIdx.x; i < kB * 32; i += 256) {
    int b = i >> 5, jl = i & 31;
    hbuf[b * kH + jbase + jl] = (__bf16)h0[b * kH + jbase + jl];
  }
  float creg[4];
#pragma unroll
  for (int k = 0; k < 4; ++k) {
    int i = threadIdx.x + k * 256;
    int b = i >> 5, jl = i & 31;
    creg[k] = c0[b * kH + jbase + jl];
  }
  unsigned bgen = grid_barrier(bar_count, bar_gen, NBLK, 0);

  // Column tile of this wave: gate g = wave/2, offset (wave&1)*16 within the
  // 32 owned columns of that gate.
  const int g = wave >> 1;
  const int nBase = g * kH + jbase + (wave & 1) * 16;      // row in Whh
  const int colL = g * 32 + (wave & 1) * 16 + (lane & 15); // LDS column

  for (int t = 0; t < kT; ++t) {
    const __bf16* hread = hbuf + (size_t)(t & 1) * kB * kH;
    __bf16* hwrite = hbuf + (size_t)((t + 1) & 1) * kB * kH;

    // Z[0:32, tile] = h @ Whh^T tile : K = 1024 in steps of 32
    v8f acc0 = v8f{}, acc1 = v8f{};
    for (int k0 = 0; k0 < kH; k0 += 32) {
      v16bf bm = load_b_wrow(Whh, kH, nBase, k0, lane);
      v16bf a0 = load_a_bf16(hread, kH, 0, k0, lane);
      v16bf a1 = load_a_bf16(hread, kH, 16, k0, lane);
      acc0 = wmma_bf16(a0, bm, acc0);
      acc1 = wmma_bf16(a1, bm, acc1);
    }
    int rowH = ((lane >> 4) & 1) * 8;
#pragma unroll
    for (int vi = 0; vi < 8; ++vi) zsm[rowH + vi][colL] = acc0[vi];
#pragma unroll
    for (int vi = 0; vi < 8; ++vi) zsm[16 + rowH + vi][colL] = acc1[vi];
    __syncthreads();

    // Fused LSTM cell + residual for owned units.
    const float* gin_t = Gin + (size_t)t * kB * kG;
    const float* xin_t = x_in + (size_t)t * kB * kH;
    float* xout_t = x_out + (size_t)t * kB * kH;
    __bf16* xoutb_t = x_out_bf + (size_t)t * kB * kH;
#pragma unroll
    for (int k = 0; k < 4; ++k) {
      int i = threadIdx.x + k * 256;
      int b = i >> 5, jl = i & 31;
      int j = jbase + jl;
      const float* gb = gin_t + (size_t)b * kG;
      float zi = zsm[b][0 * 32 + jl] + gb[0 * kH + j];
      float zf = zsm[b][1 * 32 + jl] + gb[1 * kH + j];
      float zg = zsm[b][2 * 32 + jl] + gb[2 * kH + j];
      float zo = zsm[b][3 * 32 + jl] + gb[3 * kH + j];
      float cn = sigmoidf(zf) * creg[k] + sigmoidf(zi) * tanhf(zg);
      float hn = sigmoidf(zo) * tanhf(cn);
      creg[k] = cn;
      float xr = hn + xin_t[b * kH + j];
      hwrite[b * kH + j] = (__bf16)hn;                       // publish for t+1
      xout_t[b * kH + j] = xr;                               // residual (f32)
      xoutb_t[b * kH + j] = (__bf16)xr;                      // GEMM A shadow
      if (t == kT - 1) { hT[b * kH + j] = hn; cT[b * kH + j] = cn; }
    }
    bgen = grid_barrier(bar_count, bar_gen, NBLK, bgen);     // h published
  }
}

// ---------------------------------------------------------------------------
// Kernel 5: classifier GEMM  logits[b,t,v] = Xbf[t*B+b,:] @ cls_W[v,:]+cls_b[v]
// M=4096, N=32000, K=1024 — pure-bf16 fragment loads, scattered [B,T,V] store.
// ---------------------------------------------------------------------------
__global__ void __launch_bounds__(128)
classifier_kernel(const __bf16* __restrict__ Xbf, const __bf16* __restrict__ W,
                  const float* __restrict__ bias, float* __restrict__ logits) {
  int lane = threadIdx.x & 31;
  int wave = threadIdx.x >> 5;
  int mBase = blockIdx.x * 16;
  int nBase = blockIdx.y * 256 + wave * 64;

  v8f acc[4] = {v8f{}, v8f{}, v8f{}, v8f{}};
  for (int k0 = 0; k0 < kH; k0 += 32) {
    v16bf a = load_a_bf16(Xbf, kH, mBase, k0, lane);
    __builtin_prefetch(W + (size_t)(nBase + (lane & 15)) * kH + k0 + 256, 0, 1);
#pragma unroll
    for (int j = 0; j < 4; ++j) {
      v16bf b = load_b_wrow(W, kH, nBase + j * 16, k0, lane);
      acc[j] = wmma_bf16(a, b, acc[j]);
    }
  }
  int col0 = lane & 15;
  int rbase = mBase + ((lane >> 4) & 1) * 8;
#pragma unroll
  for (int j = 0; j < 4; ++j) {
    int n = nBase + j * 16 + col0;
    float bv = bias[n];
#pragma unroll
    for (int vi = 0; vi < 8; ++vi) {
      int m = rbase + vi;                 // m = t*B + b
      int t = m >> 5, b = m & 31;
      logits[((size_t)b * kT + t) * kV + n] = acc[j][vi] + bv;
    }
  }
}

// ---------------------------------------------------------------------------
// Host orchestration (graph-capture safe: only async launches + memsetAsync)
// ---------------------------------------------------------------------------
extern "C" void kernel_launch(void* const* d_in, const int* in_sizes, int n_in,
                              void* d_out, int out_size, void* d_ws, size_t ws_size,
                              hipStream_t stream) {
  (void)in_sizes; (void)n_in; (void)out_size; (void)ws_size;
  const int*   dec  = (const int*)d_in[0];
  const float* h0   = (const float*)d_in[1];
  const float* c0   = (const float*)d_in[2];
  const float* emb  = (const float*)d_in[3];
  const float* Wih  = (const float*)d_in[4];
  const float* Whh  = (const float*)d_in[5];
  const float* bih  = (const float*)d_in[6];
  const float* bhh  = (const float*)d_in[7];
  const float* clsW = (const float*)d_in[8];
  const float* clsb = (const float*)d_in[9];
  float* out = (float*)d_out;
  char*  ws  = (char*)d_ws;

  size_t off = 0;
  auto take = [&](size_t bytes) -> char* {
    char* p = ws + off;
    off += (bytes + 255) & ~(size_t)255;
    return p;
  };
  const size_t nW = (size_t)kL * kG * kH;       // 16.78M per weight tensor
  const size_t nC = (size_t)kV * kH;            // 32.77M
  __bf16* wih_bf = (__bf16*)take(nW * 2);       // 32 MB
  __bf16* whh_bf = (__bf16*)take(nW * 2);       // 32 MB
  __bf16* cls_bf = (__bf16*)take(nC * 2);       // 64 MB
  float*  x0     = (float*)take((size_t)kM * kH * 4);    // 16 MB
  float*  x1     = (float*)take((size_t)kM * kH * 4);    // 16 MB
  __bf16* xbf0   = (__bf16*)take((size_t)kM * kH * 2);   // 8 MB
  __bf16* xbf1   = (__bf16*)take((size_t)kM * kH * 2);   // 8 MB
  float*  gin    = (float*)take((size_t)kM * kG * 4);    // 64 MB
  __bf16* hbuf   = (__bf16*)take((size_t)2 * kB * kH * 2);
  unsigned* bar  = (unsigned*)take(256);

  // 1) bf16 weight copies
  cvt_bf16_kernel<<<(unsigned)((nW + 255) / 256), 256, 0, stream>>>(Wih, wih_bf, nW);
  cvt_bf16_kernel<<<(unsigned)((nW + 255) / 256), 256, 0, stream>>>(Whh, whh_bf, nW);
  cvt_bf16_kernel<<<(unsigned)((nC + 255) / 256), 256, 0, stream>>>(clsW, cls_bf, nC);

  // 2) embedding gather, time-major [T,B,D], f32 + bf16 shadow
  embed_kernel<<<kT * kB, 256, 0, stream>>>(dec, emb, x0, xbf0);

  // 3) layer-wavefront LSTM: big parallel input-gate GEMM, then persistent
  //    recurrence kernel with grid barrier.
  float*  xin   = x0;
  float*  xout  = x1;
  __bf16* xbin  = xbf0;
  __bf16* xbout = xbf1;
  float* hT_base = out + (size_t)kB * kT * kV;
  float* cT_base = hT_base + (size_t)kL * kB * kH;
  for (int l = 0; l < kL; ++l) {
    gates_gemm_kernel<<<dim3(kM / 16, kG / 256), 128, 0, stream>>>(
        xbin, wih_bf + (size_t)l * kG * kH, bih + (size_t)l * kG,
        bhh + (size_t)l * kG, gin);
    hipMemsetAsync(bar, 0, 2 * sizeof(unsigned), stream);
    lstm_layer_kernel<<<NBLK, 256, 0, stream>>>(
        gin, whh_bf + (size_t)l * kG * kH, hbuf,
        h0 + (size_t)l * kB * kH, c0 + (size_t)l * kB * kH,
        xin, xout, xbout,
        hT_base + (size_t)l * kB * kH, cT_base + (size_t)l * kB * kH,
        bar, bar + 1);
    float* tf = xin; xin = xout; xout = tf;
    __bf16* tb = xbin; xbin = xbout; xbout = tb;
  }

  // 4) classifier head -> logits [B,T,V]
  classifier_kernel<<<dim3(kM / 16, kV / 256), 128, 0, stream>>>(
      xbin, cls_bf, clsb, out);
}
